// GCNNet2_41016937677162
// MI455X (gfx1250) — matrix operand order repaired and checked by reading, not compiled, not run on hardware
//
#include <hip/hip_runtime.h>

typedef __attribute__((ext_vector_type(16))) _Float16 v16h;
typedef __attribute__((ext_vector_type(8)))  float    v8f;

#define Dc   146      // true feature dim
#define DP   160      // padded feature dim (10 N-tiles of 16, 5 K-chunks of 32)
#define KCH  5        // K chunks of 32
#define NT   10       // N tiles of 16
#define W16_HALFS (KCH * NT * 32 * 16)   // 25600 halfs = 51200 B

// ---------------------------------------------------------------------------
// Swizzle a [146x146] f32 weight matrix into per-lane WMMA B-fragment order
// (f16, zero padded to 160x160). Index layout matches the GEMM's LDS reads:
//   w16[((kc*NT+nt)*32 + lane)*16 + e]  ==  W[k][n]
//   with (contiguous K order, same as the A loads) k = kc*32 + 16*(lane>>4) + e,
//        n = nt*16 + (lane&15)
// ---------------------------------------------------------------------------
__global__ void prep_w_kernel(const float* __restrict__ W, _Float16* __restrict__ w16) {
    int i = blockIdx.x * blockDim.x + threadIdx.x;
    if (i >= W16_HALFS) return;
    int e    = i & 15;
    int lane = (i >> 4) & 31;
    int t    = i >> 9;             // kc*NT + nt, 0..49
    int kc   = t / NT;
    int nt   = t - kc * NT;
    int half = lane >> 4;
    int k = kc * 32 + 16 * half + e;
    int n = nt * 16 + (lane & 15);
    float v = (k < Dc && n < Dc) ? W[k * Dc + n] : 0.0f;
    w16[i] = (_Float16)v;
}

// Copy x [nrows x 146] into a zero-padded [nrows x 160] buffer.
__global__ void pad_x_kernel(const float* __restrict__ x, float* __restrict__ xp,
                             int nrows) {
    long long i = (long long)blockIdx.x * blockDim.x + threadIdx.x;
    long long total = (long long)nrows * DP;
    if (i >= total) return;
    int row = (int)(i / DP);
    int col = (int)(i - (long long)row * DP);
    xp[i] = (col < Dc) ? x[(size_t)row * Dc + col] : 0.0f;
}

// ---------------------------------------------------------------------------
// C[nrows x DP] = A[nrows x DP, zero-padded] @ W(160x160, swizzled f16)
//                 (+ bias over valid cols, if bias != nullptr)
// Block: 128 threads = 4 waves. Each wave computes a 16-row x 160-col strip
// using 50 v_wmma_f32_16x16x32_f16 (5 K-chunks x 10 N-tiles). All loads are
// branchless: rows clamped (masked at store), K fully padded with zeros.
// ---------------------------------------------------------------------------
__global__ void gemm_node_kernel(const float* __restrict__ A, int nrows,
                                 const _Float16* __restrict__ w16,
                                 const float* __restrict__ bias,
                                 float* __restrict__ C) {
    __shared__ _Float16 wlds[W16_HALFS];   // 51200 B

    // Cooperative load of the swizzled weight block (16B chunks)
    const int tid = threadIdx.x;
    for (int i = tid; i < W16_HALFS / 8; i += 128)
        ((int4*)wlds)[i] = ((const int4*)w16)[i];
    __syncthreads();

    const int wave = tid >> 5;
    const int lane = tid & 31;
    const int half = lane >> 4;
    const int m0   = (blockIdx.x * 4 + wave) * 16;
    const int row  = m0 + (lane & 15);
    const int rowc = row < nrows ? row : (nrows - 1);   // clamp: loads stay in-bounds
    const float* __restrict__ arow = A + (size_t)rowc * DP;

    v8f zero = {};
    v8f acc[NT];
#pragma unroll
    for (int i = 0; i < NT; ++i) acc[i] = zero;

#pragma unroll
    for (int kc = 0; kc < KCH; ++kc) {
        const int kbase = kc * 32;
        // A fragment (contiguous K order, matching B swizzle):
        //  K = kbase + 16*half + e
        // -> one run of 16 consecutive floats per lane: 4x unconditional b128.
        const float4 f0 = *(const float4*)(arow + kbase + 16 * half);
        const float4 f1 = *(const float4*)(arow + kbase + 16 * half + 4);
        const float4 f2 = *(const float4*)(arow + kbase + 16 * half + 8);
        const float4 f3 = *(const float4*)(arow + kbase + 16 * half + 12);
        v16h a;
        a[0]  = (_Float16)f0.x; a[1]  = (_Float16)f0.y;
        a[2]  = (_Float16)f0.z; a[3]  = (_Float16)f0.w;
        a[4]  = (_Float16)f1.x; a[5]  = (_Float16)f1.y;
        a[6]  = (_Float16)f1.z; a[7]  = (_Float16)f1.w;
        a[8]  = (_Float16)f2.x; a[9]  = (_Float16)f2.y;
        a[10] = (_Float16)f2.z; a[11] = (_Float16)f2.w;
        a[12] = (_Float16)f3.x; a[13] = (_Float16)f3.y;
        a[14] = (_Float16)f3.z; a[15] = (_Float16)f3.w;

#pragma unroll
        for (int nt = 0; nt < NT; ++nt) {
            v16h b = *(const v16h*)&wlds[((kc * NT + nt) * 32 + lane) * 16];
            acc[nt] = __builtin_amdgcn_wmma_f32_16x16x32_f16(
                false, a, false, b, (short)0, acc[nt], false, false);
        }
    }

    // Store: C/D f32 layout -> element r of v8f is row m0 + r + 8*half, col nt*16+(lane&15)
#pragma unroll
    for (int nt = 0; nt < NT; ++nt) {
        int col = nt * 16 + (lane & 15);
        float bv = (bias != nullptr && col < Dc) ? bias[col] : 0.0f;
#pragma unroll
        for (int r = 0; r < 8; ++r) {
            int m = m0 + r + 8 * half;
            if (m < nrows) C[(size_t)m * DP + col] = acc[nt][r] + bv;
        }
    }
}

// ---------------------------------------------------------------------------
__global__ void zero_f32_kernel(float* __restrict__ p, long long n) {
    long long i = (long long)blockIdx.x * blockDim.x + threadIdx.x;
    if (i < n) p[i] = 0.0f;
}

// One wave per edge: agg[dst][0:146] += hw[src][0:146] via L2-resident f32 atomics
__global__ void scatter_add_kernel(const int* __restrict__ ei, int nedges,
                                   const float* __restrict__ hw,
                                   float* __restrict__ agg) {
    int edge = (blockIdx.x * blockDim.x + threadIdx.x) >> 5;
    int lane = threadIdx.x & 31;
    if (edge >= nedges) return;
    int src = ei[edge];
    int dst = ei[nedges + edge];
    const float* s = hw  + (size_t)src * DP;
    float*       d = agg + (size_t)dst * DP;
    for (int c = lane; c < Dc; c += 32)
        atomicAdd(&d[c], s[c]);
}

__global__ void bias_relu_kernel(float* __restrict__ h, const float* __restrict__ b,
                                 int nrows) {
    long long i = (long long)blockIdx.x * blockDim.x + threadIdx.x;
    long long total = (long long)nrows * Dc;
    if (i >= total) return;
    int row = (int)(i / Dc);
    int col = (int)(i - (long long)row * Dc);
    float v = h[(size_t)row * DP + col] + b[col];
    h[(size_t)row * DP + col] = v > 0.0f ? v : 0.0f;
}

// One wave per node: pool[batch[n]][c] += h[n][c]; counts[batch[n]] += 1
__global__ void pool_sum_kernel(const float* __restrict__ h, const int* __restrict__ batch,
                                float* __restrict__ pool, float* __restrict__ counts,
                                int nrows) {
    int node = (blockIdx.x * blockDim.x + threadIdx.x) >> 5;
    int lane = threadIdx.x & 31;
    if (node >= nrows) return;
    int g = batch[node];
    if (lane == 0) atomicAdd(&counts[g], 1.0f);
    const float* s = h    + (size_t)node * DP;
    float*       p = pool + (size_t)g * DP;
    for (int c = lane; c < Dc; c += 32)
        atomicAdd(&p[c], s[c]);
}

// MLP readout layer 1: hg = pool/count ; z1 = relu(hg @ W1 + b1)   (146 -> 73)
__global__ void mlp1_kernel(const float* __restrict__ pool, const float* __restrict__ counts,
                            const float* __restrict__ W1, const float* __restrict__ b1,
                            float* __restrict__ z1) {
    __shared__ float hg[Dc];
    int g = blockIdx.x;
    float inv = 1.0f / fmaxf(counts[g], 1.0f);
    for (int c = threadIdx.x; c < Dc; c += blockDim.x)
        hg[c] = pool[(size_t)g * DP + c] * inv;
    __syncthreads();
    for (int j = threadIdx.x; j < 73; j += blockDim.x) {
        float s = b1[j];
        for (int c = 0; c < Dc; ++c) s = fmaf(hg[c], W1[c * 73 + j], s);
        z1[g * 73 + j] = fmaxf(s, 0.0f);
    }
}

// 73 -> 36 with relu
__global__ void mlp2_kernel(const float* __restrict__ z1, const float* __restrict__ W2,
                            const float* __restrict__ b2, float* __restrict__ z2) {
    __shared__ float zin[73];
    int g = blockIdx.x;
    for (int c = threadIdx.x; c < 73; c += blockDim.x) zin[c] = z1[g * 73 + c];
    __syncthreads();
    for (int j = threadIdx.x; j < 36; j += blockDim.x) {
        float s = b2[j];
        for (int c = 0; c < 73; ++c) s = fmaf(zin[c], W2[c * 36 + j], s);
        z2[g * 36 + j] = fmaxf(s, 0.0f);
    }
}

// 36 -> 10, no relu
__global__ void mlp3_kernel(const float* __restrict__ z2, const float* __restrict__ W3,
                            const float* __restrict__ b3, float* __restrict__ out) {
    __shared__ float zin[36];
    int g = blockIdx.x;
    for (int c = threadIdx.x; c < 36; c += blockDim.x) zin[c] = z2[g * 36 + c];
    __syncthreads();
    for (int j = threadIdx.x; j < 10; j += blockDim.x) {
        float s = b3[j];
        for (int c = 0; c < 36; ++c) s = fmaf(zin[c], W3[c * 10 + j], s);
        out[g * 10 + j] = s;
    }
}

// ---------------------------------------------------------------------------
extern "C" void kernel_launch(void* const* d_in, const int* in_sizes, int n_in,
                              void* d_out, int out_size, void* d_ws, size_t ws_size,
                              hipStream_t stream) {
    (void)n_in; (void)ws_size;

    const float* x      = (const float*)d_in[0];
    const int*   ei     = (const int*)d_in[1];     // [2, E]
    const int*   batch  = (const int*)d_in[2];     // [N]
    const float* emb_W  = (const float*)d_in[3];
    const float* emb_b  = (const float*)d_in[4];
    const float* gcn_W  = (const float*)d_in[5];   // [L, D, D]
    const float* gcn_b  = (const float*)d_in[6];   // [L, D]
    const float* r_W1   = (const float*)d_in[7];
    const float* r_b1   = (const float*)d_in[8];
    const float* r_W2   = (const float*)d_in[9];
    const float* r_b2   = (const float*)d_in[10];
    const float* r_W3   = (const float*)d_in[11];
    const float* r_b3   = (const float*)d_in[12];
    float* out = (float*)d_out;

    const int N = in_sizes[0] / Dc;          // 100000
    const int E = in_sizes[1] / 2;           // 500000
    const int G = out_size / 10;             // 256
    const int L = in_sizes[5] / (Dc * Dc);   // 4

    // Workspace carve-out (256B aligned)
    char* ws = (char*)d_ws;
    size_t off = 0;
    auto carve = [&](size_t bytes) -> void* {
        void* p = ws + off;
        off += (bytes + 255) & ~(size_t)255;
        return p;
    };
    float*     bufh   = (float*)    carve((size_t)N * DP * sizeof(float)); // h / agg
    float*     buft   = (float*)    carve((size_t)N * DP * sizeof(float)); // x padded / hw
    _Float16*  w16    = (_Float16*) carve(W16_HALFS * sizeof(_Float16));
    float*     pool   = (float*)    carve((size_t)G * DP * sizeof(float));
    float*     counts = (float*)    carve((size_t)G * sizeof(float));
    float*     z1     = (float*)    carve((size_t)G * 73 * sizeof(float));
    float*     z2     = (float*)    carve((size_t)G * 36 * sizeof(float));

    const int gemm_blocks = (N + 63) / 64;   // 4 waves/block, 16 rows each
    const long long npad = (long long)N * DP;

    // Embedding: pad x, then h = xp @ emb_W + emb_b
    pad_x_kernel<<<(int)((npad + 255) / 256), 256, 0, stream>>>(x, buft, N);
    prep_w_kernel<<<(W16_HALFS + 255) / 256, 256, 0, stream>>>(emb_W, w16);
    gemm_node_kernel<<<gemm_blocks, 128, 0, stream>>>(buft, N, w16, emb_b, bufh);

    // GCN layers
    for (int l = 0; l < L; ++l) {
        prep_w_kernel<<<(W16_HALFS + 255) / 256, 256, 0, stream>>>(gcn_W + (size_t)l * Dc * Dc, w16);
        gemm_node_kernel<<<gemm_blocks, 128, 0, stream>>>(bufh, N, w16, nullptr, buft);
        zero_f32_kernel<<<(int)((npad + 255) / 256), 256, 0, stream>>>(bufh, npad);
        scatter_add_kernel<<<(E * 32 + 255) / 256, 256, 0, stream>>>(ei, E, buft, bufh);
        long long nelem = (long long)N * Dc;
        bias_relu_kernel<<<(int)((nelem + 255) / 256), 256, 0, stream>>>(bufh, gcn_b + (size_t)l * Dc, N);
    }

    // Global mean pool
    long long npool = (long long)G * DP;
    zero_f32_kernel<<<(int)((npool + 255) / 256), 256, 0, stream>>>(pool, npool);
    zero_f32_kernel<<<(G + 255) / 256, 256, 0, stream>>>(counts, G);
    pool_sum_kernel<<<(N * 32 + 255) / 256, 256, 0, stream>>>(bufh, batch, pool, counts, N);

    // MLP readout 146 -> 73 -> 36 -> 10
    mlp1_kernel<<<G, 128, 0, stream>>>(pool, counts, r_W1, r_b1, z1);
    mlp2_kernel<<<G, 64, 0, stream>>>(z1, r_W2, r_b2, z2);
    mlp3_kernel<<<G, 64, 0, stream>>>(z2, r_W3, r_b3, out);
}
